// InfoNCEEmpowermentEstimator_31464930410743
// MI455X (gfx1250) — matrix operand order repaired and checked by reading, not compiled, not run on hardware
//
#include <hip/hip_runtime.h>
#include <cstdint>

// ---------------------------------------------------------------------------
// InfoNCE empowerment estimator for MI455X (gfx1250, wave32, WMMA).
// Dominant cost: 524288-row [256->128->128] MLP  => bf16 v_wmma_f32_16x16x32.
// Weights staged in LDS per block; each wave computes 2 M-tiles (32 rows) so
// every LDS B-fragment feeds two WMMAs.
// ---------------------------------------------------------------------------

typedef __attribute__((ext_vector_type(16))) __bf16 v16bf;
typedef __attribute__((ext_vector_type(8)))  __bf16 v8bf;
typedef __attribute__((ext_vector_type(8)))  float  v8f;
typedef __attribute__((ext_vector_type(4)))  float  v4f;

#define B_      16384
#define K_      32
#define LATENT_ 256
#define ACT_    64
#define H_      128
#define EPS_    0.01f

#define WAVES_  4          // waves per block (block = 128 threads)

// ---------------------------------------------------------------------------
// Weight prep: fp32 [K][N] row-major  ->  bf16 [N][K] (B-fragment friendly).
// ---------------------------------------------------------------------------
__global__ __launch_bounds__(256)
void prep_weights(const float* __restrict__ aw1, const float* __restrict__ aw2,
                  const float* __restrict__ lw1, const float* __restrict__ lw2,
                  __bf16* __restrict__ aw1t, __bf16* __restrict__ aw2t,
                  __bf16* __restrict__ lw1t, __bf16* __restrict__ lw2t)
{
    int i = blockIdx.x * blockDim.x + threadIdx.x;
    const int s0 = ACT_ * H_;      // a_w1: 64x128
    const int s1 = H_ * H_;        // a_w2: 128x128
    const int s2 = LATENT_ * H_;   // l_w1: 256x128
    const int s3 = H_ * H_;        // l_w2: 128x128
    if (i < s0) { int k = i / H_, n = i % H_; aw1t[n * ACT_ + k]    = (__bf16)aw1[i]; return; }
    i -= s0;
    if (i < s1) { int k = i / H_, n = i % H_; aw2t[n * H_ + k]      = (__bf16)aw2[i]; return; }
    i -= s1;
    if (i < s2) { int k = i / H_, n = i % H_; lw1t[n * LATENT_ + k] = (__bf16)lw1[i]; return; }
    i -= s2;
    if (i < s3) { int k = i / H_, n = i % H_; lw2t[n * H_ + k]      = (__bf16)lw2[i]; }
}

// --------------------------- fragment loaders ------------------------------
// A fragment (16-bit, 16x32): lane L, row M=L&15, half h=L>>4:
//   regs v0..7 = K (32kk+16h)..+15  (contiguous)
__device__ __forceinline__ v16bf load_a_f32(const float* __restrict__ row, int c0)
{
    const v4f f0 = *(const v4f*)(row + c0);
    const v4f f1 = *(const v4f*)(row + c0 + 4);
    const v4f f2 = *(const v4f*)(row + c0 + 8);
    const v4f f3 = *(const v4f*)(row + c0 + 12);
    v16bf A;
#pragma unroll
    for (int i = 0; i < 4; ++i) {
        A[i]      = (__bf16)f0[i];
        A[4 + i]  = (__bf16)f1[i];
        A[8 + i]  = (__bf16)f2[i];
        A[12 + i] = (__bf16)f3[i];
    }
    return A;
}

__device__ __forceinline__ v16bf load_a_bf16(const __bf16* __restrict__ row)
{   // 8 bf16 at row, 8 bf16 at row+8
    const v8bf a0 = *(const v8bf*)(row);
    const v8bf a1 = *(const v8bf*)(row + 8);
    v16bf A;
#pragma unroll
    for (int i = 0; i < 8; ++i) { A[i] = a0[i]; A[8 + i] = a1[i]; }
    return A;
}

// B fragment from Wt[N][K]: lane N=L&15 reads 16 contiguous K at 32kk+16h.
__device__ __forceinline__ v16bf load_b16(const __bf16* __restrict__ p)
{
    const v8bf blo = *(const v8bf*)(p);
    const v8bf bhi = *(const v8bf*)(p + 8);
    v16bf B;
#pragma unroll
    for (int i = 0; i < 8; ++i) { B[i] = blo[i]; B[8 + i] = bhi[i]; }
    return B;
}

// --------------------------- per-tile epilogue -----------------------------
// MODE 0: logits[r][0] (b = r) | MODE 1: logits[r>>5][1+(r&31)] | MODE 2: emb rows
template<int MODE>
__device__ __forceinline__ void epilogue_tile(const v8f (&c2)[8], int r0t, int lane,
                                              const float* __restrict__ b2,
                                              const float* __restrict__ emb_a,
                                              float* __restrict__ logits,
                                              float* __restrict__ emb_out,
                                              float invt)
{
    const int half = lane >> 4;
    const int mrow = lane & 15;
    if (MODE == 2) {
#pragma unroll
        for (int nt = 0; nt < 8; ++nt) {
            const float bias = b2[nt * 16 + mrow];
#pragma unroll
            for (int e = 0; e < 8; ++e)
                emb_out[(long)(r0t + half * 8 + e) * H_ + nt * 16 + mrow] = c2[nt][e] + bias;
        }
    } else {
        float p[8] = {};
#pragma unroll
        for (int nt = 0; nt < 8; ++nt) {
            const float bias = b2[nt * 16 + mrow];
            if (MODE == 1) {
                const float av = emb_a[(long)(r0t >> 5) * H_ + nt * 16 + mrow];
#pragma unroll
                for (int e = 0; e < 8; ++e) p[e] += (c2[nt][e] + bias) * av;
            } else {
#pragma unroll
                for (int e = 0; e < 8; ++e) {
                    const float av = emb_a[(long)(r0t + half * 8 + e) * H_ + nt * 16 + mrow];
                    p[e] += (c2[nt][e] + bias) * av;
                }
            }
        }
        // reduce across the 16-lane N group (wave32 shuffles)
#pragma unroll
        for (int e = 0; e < 8; ++e) {
            p[e] += __shfl_xor(p[e], 1, 32);
            p[e] += __shfl_xor(p[e], 2, 32);
            p[e] += __shfl_xor(p[e], 4, 32);
            p[e] += __shfl_xor(p[e], 8, 32);
        }
        if (mrow == 0) {
#pragma unroll
            for (int e = 0; e < 8; ++e) {
                float lg = p[e] * invt;
                lg = fminf(fmaxf(lg, -20.0f), 20.0f);
                const int rr = r0t + half * 8 + e;
                long addr;
                if (MODE == 0) addr = (long)rr * 33;                       // col 0
                else           addr = (long)(rr >> 5) * 33 + 1 + (rr & 31);
                logits[addr] = lg;
            }
        }
    }
}

// ---------------------------------------------------------------------------
// Fused 2-layer MLP via WMMA. Block = 128 threads = 4 waves; weights in LDS;
// each wave owns 2 consecutive 16-row M tiles (32 rows).
// ---------------------------------------------------------------------------
template<int KTOT1, int MODE>
__global__ __launch_bounds__(128)
void mlp_wmma_kernel(const float* __restrict__ xsrc,
                     const int*   __restrict__ neg_idx,
                     const __bf16* __restrict__ w1t,   // [H][KTOT1] bf16
                     const float* __restrict__ b1,
                     const __bf16* __restrict__ w2t,   // [H][H] bf16
                     const float* __restrict__ b2,
                     const float* __restrict__ emb_a,  // modes 0,1
                     float* __restrict__ logits,       // modes 0,1
                     float* __restrict__ emb_out,      // mode 2
                     const float* __restrict__ temp_ptr)
{
    __shared__ __align__(16) __bf16 w1s[H_ * KTOT1];            // 16/64 KB
    __shared__ __align__(16) __bf16 w2s[H_ * H_];               // 32 KB
    __shared__ __align__(16) __bf16 hbuf[WAVES_][2][16 * H_];   // 32 KB

    const int tid  = threadIdx.x;
    const int lane = tid & 31;
    const int wave = tid >> 5;
    const int half = lane >> 4;
    const int mrow = lane & 15;
    const int r0   = (blockIdx.x * WAVES_ + wave) * 32;   // 32 rows per wave

    // ---- cooperative weight stage: global bf16 -> LDS ----
#pragma unroll
    for (int i = tid; i < H_ * KTOT1 / 8; i += 128)
        ((v8bf*)w1s)[i] = ((const v8bf*)w1t)[i];
#pragma unroll
    for (int i = tid; i < H_ * H_ / 8; i += 128)
        ((v8bf*)w2s)[i] = ((const v8bf*)w2t)[i];

    // source row pointers for the two tiles (per-lane rows)
    const float* xrow0;
    const float* xrow1;
    if (MODE == 1) {
        xrow0 = xsrc + (long)neg_idx[r0 + mrow] * LATENT_;
        xrow1 = xsrc + (long)neg_idx[r0 + 16 + mrow] * LATENT_;
    } else {
        xrow0 = xsrc + (long)(r0 + mrow) * KTOT1;
        xrow1 = xsrc + (long)(r0 + 16 + mrow) * KTOT1;
    }
    __syncthreads();

    // ----------------- layer 1: [2 x 16 x KTOT1] @ [KTOT1 x 128] -----------
    v8f c1a[8] = {}, c1b[8] = {};
#pragma unroll
    for (int kk = 0; kk < KTOT1 / 32; ++kk) {
        const int c0 = kk * 32 + half * 16;
        const v16bf A0 = load_a_f32(xrow0, c0);
        const v16bf A1 = load_a_f32(xrow1, c0);
#pragma unroll
        for (int nt = 0; nt < 8; ++nt) {
            const v16bf Bf = load_b16(w1s + (nt * 16 + mrow) * KTOT1 + kk * 32 + half * 16);
            c1a[nt] = __builtin_amdgcn_wmma_f32_16x16x32_bf16(
                false, A0, false, Bf, (short)0, c1a[nt], false, false);
            c1b[nt] = __builtin_amdgcn_wmma_f32_16x16x32_bf16(
                false, A1, false, Bf, (short)0, c1b[nt], false, false);
        }
    }

    // bias + ReLU, restage C-frag -> A-frag layout via LDS (bf16)
#pragma unroll
    for (int nt = 0; nt < 8; ++nt) {
        const float bias = b1[nt * 16 + mrow];
#pragma unroll
        for (int e = 0; e < 8; ++e) {
            float va = c1a[nt][e] + bias;
            float vb = c1b[nt][e] + bias;
            va = va > 0.0f ? va : 0.0f;
            vb = vb > 0.0f ? vb : 0.0f;
            hbuf[wave][0][(half * 8 + e) * H_ + nt * 16 + mrow] = (__bf16)va;
            hbuf[wave][1][(half * 8 + e) * H_ + nt * 16 + mrow] = (__bf16)vb;
        }
    }
    __syncthreads();

    // ----------------- layer 2: [2 x 16 x 128] @ [128 x 128] ---------------
    v8f c2a[8] = {}, c2b[8] = {};
#pragma unroll
    for (int kk = 0; kk < 4; ++kk) {
        const v16bf A0 = load_a_bf16(&hbuf[wave][0][mrow * H_ + kk * 32 + half * 16]);
        const v16bf A1 = load_a_bf16(&hbuf[wave][1][mrow * H_ + kk * 32 + half * 16]);
#pragma unroll
        for (int nt = 0; nt < 8; ++nt) {
            const v16bf Bf = load_b16(w2s + (nt * 16 + mrow) * H_ + kk * 32 + half * 16);
            c2a[nt] = __builtin_amdgcn_wmma_f32_16x16x32_bf16(
                false, A0, false, Bf, (short)0, c2a[nt], false, false);
            c2b[nt] = __builtin_amdgcn_wmma_f32_16x16x32_bf16(
                false, A1, false, Bf, (short)0, c2b[nt], false, false);
        }
    }

    float invt = 0.0f;
    if (MODE != 2) {
        float t = temp_ptr[0];
        if (!__builtin_isfinite(t)) t = 0.1f;
        t = fminf(fmaxf(t, 0.01f), 10.0f);
        invt = 1.0f / t;
    }
    epilogue_tile<MODE>(c2a, r0,      lane, b2, emb_a, logits, emb_out, invt);
    epilogue_tile<MODE>(c2b, r0 + 16, lane, b2, emb_a, logits, emb_out, invt);
}

// ---------------------------------------------------------------------------
// Per-sample log-softmax over 33 logits + label smoothing + clip.
// ---------------------------------------------------------------------------
__global__ __launch_bounds__(256)
void loss_kernel(const float* __restrict__ logits, float* __restrict__ out)
{
    const int b = blockIdx.x * blockDim.x + threadIdx.x;
    if (b >= B_) return;
    const float* l = logits + (long)b * 33;
    float m = -1e30f;
#pragma unroll
    for (int i = 0; i < 33; ++i) m = fmaxf(m, l[i]);
    float s = 0.0f, sum = 0.0f;
#pragma unroll
    for (int i = 0; i < 33; ++i) { s += __expf(l[i] - m); sum += l[i]; }
    const float lse      = m + __logf(s);
    const float logp0    = l[0] - lse;
    const float meanlogp = sum * (1.0f / 33.0f) - lse;
    float loss = -(1.0f - EPS_) * logp0 - EPS_ * meanlogp;
    if (!__builtin_isfinite(loss)) loss = 0.0f;
    loss = fminf(fmaxf(loss, -10.0f), 10.0f);
    out[b] = -loss;
}

// ---------------------------------------------------------------------------
extern "C" void kernel_launch(void* const* d_in, const int* in_sizes, int n_in,
                              void* d_out, int out_size, void* d_ws, size_t ws_size,
                              hipStream_t stream)
{
    (void)in_sizes; (void)n_in; (void)out_size; (void)ws_size;

    const float* action  = (const float*)d_in[0];
    const float* latent  = (const float*)d_in[1];
    const float* queue   = (const float*)d_in[2];
    const int*   neg_idx = (const int*)  d_in[3];
    const float* a_w1    = (const float*)d_in[4];
    const float* a_b1    = (const float*)d_in[5];
    const float* a_w2    = (const float*)d_in[6];
    const float* a_b2    = (const float*)d_in[7];
    const float* l_w1    = (const float*)d_in[8];
    const float* l_b1    = (const float*)d_in[9];
    const float* l_w2    = (const float*)d_in[10];
    const float* l_b2    = (const float*)d_in[11];
    const float* temp    = (const float*)d_in[12];
    float* out = (float*)d_out;

    // workspace layout
    char*  ws     = (char*)d_ws;
    float* emb_a  = (float*)ws;                                   // B*128 f32 (8 MB)
    size_t off    = (size_t)B_ * H_ * 4;
    float* logits = (float*)(ws + off);                           // B*33 f32 (2.2 MB)
    off += (size_t)B_ * 33 * 4;
    off  = (off + 255) & ~(size_t)255;
    __bf16* aw1t = (__bf16*)(ws + off); off += (size_t)ACT_ * H_ * 2;
    __bf16* aw2t = (__bf16*)(ws + off); off += (size_t)H_ * H_ * 2;
    __bf16* lw1t = (__bf16*)(ws + off); off += (size_t)LATENT_ * H_ * 2;
    __bf16* lw2t = (__bf16*)(ws + off);

    // 1) weights -> transposed bf16 (73728 elems total)
    prep_weights<<<288, 256, 0, stream>>>(a_w1, a_w2, l_w1, l_w2,
                                          aw1t, aw2t, lw1t, lw2t);

    // 2) emb_a = aMLP(action): 16384 rows / (4 waves * 32 rows) = 128 blocks
    mlp_wmma_kernel<ACT_, 2><<<128, 128, 0, stream>>>(
        action, nullptr, aw1t, a_b1, aw2t, a_b2, nullptr, nullptr, emb_a, temp);

    // 3) emb_l = lMLP(latent) fused with logits[:,0]: 128 blocks
    mlp_wmma_kernel<LATENT_, 0><<<128, 128, 0, stream>>>(
        latent, nullptr, lw1t, l_b1, lw2t, l_b2, emb_a, logits, nullptr, temp);

    // 4) neg = lMLP(queue[neg_idx]) fused with logits[:,1:33]:
    //    524288 rows / (4 waves * 32 rows) = 4096 blocks
    mlp_wmma_kernel<LATENT_, 1><<<4096, 128, 0, stream>>>(
        queue, neg_idx, lw1t, l_b1, lw2t, l_b2, emb_a, logits, nullptr, temp);

    // 5) loss per sample
    loss_kernel<<<(B_ + 255) / 256, 256, 0, stream>>>(logits, out);
}
